// GRU_69475390980659
// MI455X (gfx1250) — matrix operand order repaired and checked by reading, not compiled, not run on hardware
//
#include <hip/hip_runtime.h>

typedef unsigned short u16;
typedef __attribute__((ext_vector_type(16))) __bf16 v16bf;
typedef __attribute__((ext_vector_type(8)))  __bf16 v8bf;
typedef __attribute__((ext_vector_type(8)))  float  v8f;

#define GRU_B 64
#define GRU_T 512
#define GRU_I 512
#define GRU_H 1024
#define GRU_K (GRU_H + GRU_I)   // 1536; H boundary is a multiple of 32
#define NBLK  64                // persistent blocks; block b owns hcols [16b,16b+16)
#define NTHR  128               // 4 waves per block (one per C-tile)

// LDS layout (all offsets in u16 units unless noted):
//   weights: [gate 0..2][row 0..15][K padded to KP]
//   x tiles: two buffers of [64 rows][XROW]
#define KP       1544                       // 3088 B row stride (772 dw % 64 = 4)
#define WOFF_U16 (3 * 16 * KP)              // 74112 u16 = 148224 B
#define XROW     520                        // 1040 B row stride (260 dw % 64 = 4)
#define XBUF_U16 (64 * XROW)                // 33280 u16 = 66560 B
#define LDS_BYTES ((WOFF_U16 + 2 * XBUF_U16) * 2)   // 281,344 B <= 320 KB

// ---------- helpers ----------

__device__ __forceinline__ u16 f2bf(float f) {
  union { float f; unsigned u; } v; v.f = f;
  unsigned r = v.u + 0x7FFFu + ((v.u >> 16) & 1u);   // round-to-nearest-even
  return (u16)(r >> 16);
}

__device__ __forceinline__ float sigmoid_f(float x) {
  return 1.0f / (1.0f + __expf(-x));
}

__device__ __forceinline__ v8f wmma_bf16(v16bf a, v16bf b, v8f c) {
  return __builtin_amdgcn_wmma_f32_16x16x32_bf16(false, a, false, b, (short)0, c,
                                                 false, false);
}

// fragment = two contiguous 16B halves (ISA 7.12.2 per-lane layout)
__device__ __forceinline__ v16bf load_frag_2x8(const u16* p) {
  v8bf a0 = *(const v8bf*)(p);
  v8bf a1 = *(const v8bf*)(p + 16);
  return __builtin_shufflevector(a0, a1, 0, 1, 2, 3, 4, 5, 6, 7,
                                 8, 9, 10, 11, 12, 13, 14, 15);
}

// CDNA5 async global->LDS copy (ASYNCcnt-tracked, no VGPR data round trip).
// ldsoff: per-lane byte offset into the workgroup LDS allocation (VDST),
// gp: per-lane 64-bit global address (VADDR), GV mode.
__device__ __forceinline__ void async_g2l_b128(unsigned ldsoff, const void* gp) {
  asm volatile("global_load_async_to_lds_b128 %0, %1, off"
               :: "v"(ldsoff), "v"(gp) : "memory");
}
__device__ __forceinline__ void wait_asynccnt0() {
  asm volatile("s_wait_asynccnt 0x0" ::: "memory");
}

// software device-wide barrier: single L2 atomic counter, monotone targets
__device__ __forceinline__ void gbar(volatile unsigned* cnt, unsigned target) {
  __syncthreads();
  __threadfence();                      // release: make our stores visible
  if (threadIdx.x == 0) {
    atomicAdd((unsigned*)cnt, 1u);
    while (*cnt < target) __builtin_amdgcn_s_sleep(2);
  }
  __syncthreads();
  __threadfence();                      // acquire: drop stale cached lines
}

// ---------- setup kernels ----------

__global__ __launch_bounds__(256) void k_cvt_bf16(const float* __restrict__ src,
                                                  u16* __restrict__ dst, int n) {
  int i = blockIdx.x * blockDim.x + threadIdx.x;
  int stride = blockDim.x * gridDim.x;
  for (; i < n; i += stride) dst[i] = f2bf(src[i]);
}

__global__ void k_zero_u32(unsigned* __restrict__ p) {
  if (threadIdx.x == 0 && blockIdx.x == 0) *p = 0u;
}

// ---------- persistent GRU: all 512 timesteps in one launch ----------
// 64 blocks x 128 threads. Wave wid = mtile (batch tile 0..3); block owns
// output columns [16*blk, 16*blk+16). Weights live in LDS for the whole run,
// x_t tiles are double-buffered in LDS via async copies, h and z stay in
// registers, only the bf16 GEMM operands (h, r*h) round-trip global/L2.
__global__ __launch_bounds__(NTHR) void k_gru_persistent(
    const u16* __restrict__ xb,                         // [B,T,I] bf16
    const float* __restrict__ Uz, const float* __restrict__ Wz,
    const float* __restrict__ Ur, const float* __restrict__ Wr,
    const float* __restrict__ Uh, const float* __restrict__ Wh,
    const float* __restrict__ bz, const float* __restrict__ br,
    const float* __restrict__ bh, const float* __restrict__ h0,
    u16* __restrict__ hbf,                              // [B,H] bf16 (A operand)
    u16* __restrict__ rhb,                              // [B,H] bf16 (A operand)
    float* __restrict__ out, float* __restrict__ h_last,
    unsigned* __restrict__ barcnt) {
  extern __shared__ u16 lds[];

  const int lane  = threadIdx.x & 31;
  const int mtile = threadIdx.x >> 5;                 // wave id = batch tile
  const int hi    = lane >> 4;
  const int l16   = lane & 15;
  const int mrow  = mtile * 16 + l16;                 // A row (batch)
  const int ncol  = blockIdx.x * 16 + l16;            // B row / C col
  const int koff_a = hi * 8;
  const int koff_b = hi * 8;

  // ---- one-time: this block's weight slice -> LDS (f32 -> bf16 on the fly)
  {
    const int ncb = blockIdx.x * 16;
    const int CH = GRU_K / 4;
    for (int c = threadIdx.x; c < 3 * 16 * CH; c += blockDim.x) {
      const int g   = c / (16 * CH);
      const int rem = c % (16 * CH);
      const int r0  = rem / CH;
      const int k   = (rem % CH) * 4;
      const int n   = ncb + r0;
      const float* U = (g == 0) ? Uz : (g == 1) ? Ur : Uh;
      const float* W = (g == 0) ? Wz : (g == 1) ? Wr : Wh;
      const float4 s = (k < GRU_H)
          ? *(const float4*)(U + (size_t)n * GRU_H + k)
          : *(const float4*)(W + (size_t)n * GRU_I + (k - GRU_H));
      u16 q[4] = { f2bf(s.x), f2bf(s.y), f2bf(s.z), f2bf(s.w) };
      *(uint2*)(&lds[(size_t)(g * 16 + r0) * KP + k]) = *(const uint2*)q;
    }
  }

  // ---- stage x_0 into LDS buffer 0 (async bulk copy) ----
  // 64 rows x 512 bf16 = 4096 16B chunks; 32 async issues per wave (<= 63).
  auto stage_x = [&](int t, int buf) {
    const unsigned base = (unsigned)(WOFF_U16 + buf * XBUF_U16) * 2u;
    for (int c = threadIdx.x; c < 64 * 64; c += NTHR) {
      const int row = c >> 6;           // batch row
      const int ch  = c & 63;           // 16B chunk within row
      const u16* g = xb + ((size_t)row * GRU_T + t) * GRU_I + ch * 8;
      async_g2l_b128(base + (unsigned)(row * XROW + ch * 8) * 2u, g);
    }
  };
  stage_x(0, 0);
  wait_asynccnt0();
  __syncthreads();

  // per-lane LDS row bases for this wave's B fragments
  const u16* ldsZ = &lds[(size_t)(0 * 16 + l16) * KP];
  const u16* ldsR = &lds[(size_t)(1 * 16 + l16) * KP];
  const u16* ldsH = &lds[(size_t)(2 * 16 + l16) * KP];

  // ---- init: h state in registers; publish bf16 h for the GEMMs ----
  float hreg[8];
  #pragma unroll
  for (int j = 0; j < 8; ++j) {
    const int brow = mtile * 16 + hi * 8 + j;           // C layout row
    const size_t idx = (size_t)brow * GRU_H + ncol;
    const float v = h0[idx];
    hreg[j] = v;
    hbf[idx] = f2bf(v);
  }
  const float zb_ = bz[ncol];
  const float rb_ = br[ncol];
  const float hb_ = bh[ncol];

  unsigned nbar = 0;
  gbar(barcnt, ++nbar * NBLK);   // h published before step 0

  for (int t = 0; t < GRU_T; ++t) {
    // prefetch next timestep's x tile into the other buffer (overlapped)
    if (t + 1 < GRU_T) stage_x(t + 1, (t + 1) & 1);

    const u16* xlds = &lds[WOFF_U16 + (t & 1) * XBUF_U16 + mrow * XROW];

    // ---- phase 1: z and r gates. A = [h | x_t], B = [U^T;W^T] (LDS) ----
    v8f accz = {};
    v8f accr = {};
    {
      const u16* hrow = hbf + (size_t)mrow * GRU_H;
      #pragma unroll 4
      for (int kb = 0; kb < GRU_H; kb += 32) {          // A from global h
        v16bf a = load_frag_2x8(hrow + kb + koff_a);
        accz = wmma_bf16(a, load_frag_2x8(ldsZ + kb + koff_b), accz);
        accr = wmma_bf16(a, load_frag_2x8(ldsR + kb + koff_b), accr);
      }
      #pragma unroll 4
      for (int kb = 0; kb < GRU_I; kb += 32) {          // A from LDS x tile
        v16bf a = load_frag_2x8(xlds + kb + koff_a);
        accz = wmma_bf16(a, load_frag_2x8(ldsZ + GRU_H + kb + koff_b), accz);
        accr = wmma_bf16(a, load_frag_2x8(ldsR + GRU_H + kb + koff_b), accr);
      }
    }
    float zreg[8];
    #pragma unroll
    for (int j = 0; j < 8; ++j) {
      const int brow = mtile * 16 + hi * 8 + j;
      const size_t idx = (size_t)brow * GRU_H + ncol;
      zreg[j] = sigmoid_f(accz[j] + zb_);
      const float r = sigmoid_f(accr[j] + rb_);
      rhb[idx] = f2bf(r * hreg[j]);                    // publish r*h operand
    }
    gbar(barcnt, ++nbar * NBLK);

    // ---- phase 2: candidate + state update. A = [r*h | x_t] ----
    v8f acc = {};
    {
      const u16* rrow = rhb + (size_t)mrow * GRU_H;
      #pragma unroll 4
      for (int kb = 0; kb < GRU_H; kb += 32) {
        v16bf a = load_frag_2x8(rrow + kb + koff_a);
        acc = wmma_bf16(a, load_frag_2x8(ldsH + kb + koff_b), acc);
      }
      #pragma unroll 4
      for (int kb = 0; kb < GRU_I; kb += 32) {
        v16bf a = load_frag_2x8(xlds + kb + koff_a);
        acc = wmma_bf16(a, load_frag_2x8(ldsH + GRU_H + kb + koff_b), acc);
      }
    }
    #pragma unroll
    for (int j = 0; j < 8; ++j) {
      const int brow = mtile * 16 + hi * 8 + j;
      const size_t idx = (size_t)brow * GRU_H + ncol;
      const float ht = tanhf(acc[j] + hb_);
      const float hn = hreg[j] + zreg[j] * (ht - hreg[j]);  // (1-z)h + z*h~
      hreg[j] = hn;
      hbf[idx] = f2bf(hn);                             // next step's A operand
      out[((size_t)brow * GRU_T + t) * GRU_H + ncol] = hn;
    }

    wait_asynccnt0();              // x_{t+1} tile resident before next step
    gbar(barcnt, ++nbar * NBLK);   // (its __syncthreads gives cross-wave vis)
  }

  // ---- h_last straight from registers ----
  #pragma unroll
  for (int j = 0; j < 8; ++j) {
    const int brow = mtile * 16 + hi * 8 + j;
    h_last[(size_t)brow * GRU_H + ncol] = hreg[j];
  }
}

// ---------- host ----------

extern "C" void kernel_launch(void* const* d_in, const int* in_sizes, int n_in,
                              void* d_out, int out_size, void* d_ws, size_t ws_size,
                              hipStream_t stream) {
  const float* x  = (const float*)d_in[0];
  const float* h0 = (const float*)d_in[1];
  const float* Wz = (const float*)d_in[2];
  const float* bz = (const float*)d_in[3];
  const float* Uz = (const float*)d_in[4];
  const float* Wr = (const float*)d_in[5];
  const float* br = (const float*)d_in[6];
  const float* Ur = (const float*)d_in[7];
  const float* Wh = (const float*)d_in[8];
  const float* bh = (const float*)d_in[9];
  const float* Uh = (const float*)d_in[10];

  float* out    = (float*)d_out;                       // [B,T,H]
  float* h_last = out + (size_t)GRU_B * GRU_T * GRU_H; // [B,H]

  char* base = (char*)d_ws;
  size_t off = 0;
  auto carve = [&](size_t bytes) -> void* {
    void* r = base + off;
    off += (bytes + 255) & ~(size_t)255;
    return r;
  };
  u16* xb        = (u16*)carve((size_t)GRU_B * GRU_T * GRU_I * sizeof(u16));
  u16* hbf       = (u16*)carve((size_t)GRU_B * GRU_H * sizeof(u16));
  u16* rhb       = (u16*)carve((size_t)GRU_B * GRU_H * sizeof(u16));
  unsigned* barc = (unsigned*)carve(256);
  (void)ws_size; (void)in_sizes; (void)n_in; (void)out_size;

  k_cvt_bf16<<<4096, 256, 0, stream>>>(x, xb, GRU_B * GRU_T * GRU_I);
  k_zero_u32<<<1, 64, 0, stream>>>(barc);
  k_gru_persistent<<<NBLK, NTHR, LDS_BYTES, stream>>>(
      xb, Uz, Wz, Ur, Wr, Uh, Wh, bz, br, bh, h0,
      hbf, rhb, out, h_last, barc);
}
